// MultiTaskWeightGenerator_86912958202450
// MI455X (gfx1250) — matrix operand order repaired and checked by reading, not compiled, not run on hardware
//
#include <hip/hip_runtime.h>
#include <stdint.h>

typedef __attribute__((ext_vector_type(16))) __bf16        v16bf;
typedef __attribute__((ext_vector_type(8)))  float         v8f;
typedef __attribute__((ext_vector_type(8)))  unsigned int  v8u;

// ---------- helpers ----------
// f32x2 -> packed bf16x2 via CDNA5 V_FMA_MIXLO/MIXHI_BF16 (2 VALU ops, RNE convert).
__device__ __forceinline__ unsigned int pack_bf16(float x, float y) {
    unsigned int d = 0;
    asm("v_fma_mixlo_bf16 %0, %1, 1.0, 0\n\t"
        "v_fma_mixhi_bf16 %0, %2, 1.0, 0"
        : "+v"(d)
        : "v"(x), "v"(y));
    return d;
}
__device__ __forceinline__ unsigned short bf16_1(float x) {
    unsigned int d = 0;
    asm("v_fma_mixlo_bf16 %0, %1, 1.0, 0" : "+v"(d) : "v"(x));
    return (unsigned short)d;
}
__device__ __forceinline__ v8f wmma_bf16(v16bf a, v16bf b, v8f c) {
    return __builtin_amdgcn_wmma_f32_16x16x32_bf16(false, a, false, b, (short)0, c, false, false);
}
// Load one 16-bit operand fragment: two contiguous 16B LDS reads (ds_load_b128).
// A-layout (16x32): lane m holds pairs at elements {8h..8h+7} and {16+8h..16+8h+7}.
__device__ __forceinline__ v16bf load_frag_a(const unsigned short* rowp, int half) {
    const uint4 lo = *reinterpret_cast<const uint4*>(rowp + 8 * half);
    const uint4 hi = *reinterpret_cast<const uint4*>(rowp + 16 + 8 * half);
    v8u u; u[0] = lo.x; u[1] = lo.y; u[2] = lo.z; u[3] = lo.w;
           u[4] = hi.x; u[5] = hi.y; u[6] = hi.z; u[7] = hi.w;
    return __builtin_bit_cast(v16bf, u);
}
// B-layout (32x16): lane n holds elements {8h..8h+7} and {16+8h..16+8h+7} of column n.
__device__ __forceinline__ v16bf load_frag_b(const unsigned short* rowp, int half) {
    const uint4 lo = *reinterpret_cast<const uint4*>(rowp + 8 * half);
    const uint4 hi = *reinterpret_cast<const uint4*>(rowp + 16 + 8 * half);
    v8u u; u[0] = lo.x; u[1] = lo.y; u[2] = lo.z; u[3] = lo.w;
           u[4] = hi.x; u[5] = hi.y; u[6] = hi.z; u[7] = hi.w;
    return __builtin_bit_cast(v16bf, u);
}

// ---------- kernel 1: stable counting sort of samples by task ----------
#define NB 4096
#define NT 8

__global__ __launch_bounds__(256) void bucket_kernel(const int* __restrict__ task_ids,
                                                     int* __restrict__ order,
                                                     int* __restrict__ count,
                                                     int* __restrict__ off) {
    __shared__ int lcnt[NT][256];
    __shared__ int tot_sh[NT];
    __shared__ int off_sh[NT];
    const int tid  = threadIdx.x;
    const int base = tid * 16;
    int tl[16];
#pragma unroll
    for (int j = 0; j < 16; ++j) tl[j] = task_ids[base + j];
#pragma unroll
    for (int t = 0; t < NT; ++t) {
        int cc = 0;
#pragma unroll
        for (int j = 0; j < 16; ++j) cc += (tl[j] == t) ? 1 : 0;
        lcnt[t][tid] = cc;
    }
    __syncthreads();
    if (tid < NT) {                      // exclusive scan over the 256 thread-counts
        int run = 0;
        for (int i = 0; i < 256; ++i) { int v = lcnt[tid][i]; lcnt[tid][i] = run; run += v; }
        tot_sh[tid] = run;
        count[tid]  = run;
    }
    __syncthreads();
    if (tid == 0) {
        int o = 0;
        for (int t = 0; t < NT; ++t) { off_sh[t] = o; off[t] = o; o += tot_sh[t]; }
    }
    __syncthreads();
#pragma unroll
    for (int t = 0; t < NT; ++t) {
        int pos = off_sh[t] + lcnt[t][tid];
#pragma unroll
        for (int j = 0; j < 16; ++j) {
            if (tl[j] == t) { order[pos] = base + j; ++pos; }
        }
    }
}

// ---------- kernel 2: fused per-(task, 32-sample tile) pipeline ----------
// GEMM1(4096->512,bf16 wmma) -> +b1 -> LN -> ReLU -> GEMM2(512->256) -> +b2 -> LN -> ReLU
// -> concat(task_emb) -> weight-net(512->4) -> softmax -> scatter
__global__ __launch_bounds__(256) void fused_kernel(
    const float* __restrict__ f0, const float* __restrict__ f1,
    const float* __restrict__ f2, const float* __restrict__ f3,
    const float* __restrict__ W1, const float* __restrict__ b1,
    const float* __restrict__ g1, const float* __restrict__ be1,
    const float* __restrict__ W2, const float* __restrict__ b2,
    const float* __restrict__ g2, const float* __restrict__ be2,
    const float* __restrict__ temb, const float* __restrict__ Ww,
    const float* __restrict__ bw,
    const int* __restrict__ order, const int* __restrict__ count,
    const int* __restrict__ off, float* __restrict__ out) {

    const int t    = blockIdx.y;
    const int tile = blockIdx.x;
    const int cnt  = count[t];
    if (tile * 32 >= cnt) return;
    int n_valid = cnt - tile * 32;
    if (n_valid > 32) n_valid = 32;

    // 80B / 1040B row strides: multiple of 16B (b128-aligned), bank-conflict free.
    __shared__ int idx_sh[32];
    __shared__ __align__(16) unsigned short a_lds[32][40];   // A tile, 32 rows x 32 k (+pad)
    __shared__ __align__(16) unsigned short bt_lds[512][40]; // B^T tile, n x k (+pad)
    __shared__ __align__(16) unsigned short h1bf[32][520];   // LN1 out, bf16 row-major
    __shared__ float psum[32][8];
    __shared__ float psq[32][8];
    __shared__ float meanv[32];
    __shared__ float rstdv[32];
    __shared__ float wwl[512][4];
    __shared__ float tel[256];
    __shared__ float bwl[4];
    float (*h2n)[264] = reinterpret_cast<float (*)[264]>(&bt_lds[0][0]); // alias (GEMM2 done)

    const int tid  = threadIdx.x;
    const int wv   = tid >> 5;
    const int lane = tid & 31;
    const int half = lane >> 4;
    const int lq   = lane & 15;

    if (tid < 32) {
        int g = off[t] + tile * 32 + tid;
        idx_sh[tid] = (tid < n_valid) ? order[g] : order[off[t]];
    }
    __syncthreads();

    // hoisted A-staging coordinates: one register-resident sample index per thread
    const int a_row = tid >> 3;
    const int a_c0  = (tid & 7) * 4;
    const int ridx  = idx_sh[a_row];

    const v8f zero8 = {0.f, 0.f, 0.f, 0.f, 0.f, 0.f, 0.f, 0.f};

    // ---------------- GEMM1: [32 x 4096] x [4096 x 512] ----------------
    v8f acc1[2][4];
#pragma unroll
    for (int mt = 0; mt < 2; ++mt)
#pragma unroll
        for (int nt = 0; nt < 4; ++nt) acc1[mt][nt] = zero8;

    // 4 modality segments (1024 cols each) so the source pointer is loop-invariant
#pragma unroll 1
    for (int seg = 0; seg < 4; ++seg) {
        const float* fsrc = (seg == 0) ? f0 : (seg == 1) ? f1 : (seg == 2) ? f2 : f3;
        const float* asrc = fsrc + (size_t)ridx * 1024 + a_c0;            // this thread's A source
        const float* wseg = W1 + ((size_t)t * 4096 + (size_t)seg * 1024) * 512;

#pragma unroll 1
        for (int kk = 0; kk < 1024; kk += 32) {
            // stage A: 32 rows x 32 cols f32 -> bf16 LDS (one b64 store per thread)
            {
                const float4 v = *reinterpret_cast<const float4*>(asrc + kk);
                uint2 u;
                u.x = pack_bf16(v.x, v.y);
                u.y = pack_bf16(v.z, v.w);
                *reinterpret_cast<uint2*>(&a_lds[a_row][a_c0]) = u;
            }
            // stage B: W1 rows kk..kk+31 -> transposed bf16, 4 k per b64 store
            {
                const float* wbase = wseg + (size_t)kk * 512;
                if (kk + 32 < 1024)   // prefetch next chunk (64KB) into cache
                    __builtin_prefetch(wbase + 32 * 512 + tid * 64, 0, 3);
#pragma unroll
                for (int it = 0; it < 4; ++it) {
                    const int e  = it * 256 + tid;
                    const int q  = e >> 7;          // 0..7 -> d0 = 4q
                    const int h0 = (e & 127) * 4;
                    const int d0 = 4 * q;
                    const float* s0 = wbase + (size_t)d0 * 512 + h0;
                    const float4 r0 = *reinterpret_cast<const float4*>(s0);
                    const float4 r1 = *reinterpret_cast<const float4*>(s0 + 512);
                    const float4 r2 = *reinterpret_cast<const float4*>(s0 + 1024);
                    const float4 r3 = *reinterpret_cast<const float4*>(s0 + 1536);
                    uint2 u;
                    u.x = pack_bf16(r0.x, r1.x); u.y = pack_bf16(r2.x, r3.x);
                    *reinterpret_cast<uint2*>(&bt_lds[h0 + 0][d0]) = u;
                    u.x = pack_bf16(r0.y, r1.y); u.y = pack_bf16(r2.y, r3.y);
                    *reinterpret_cast<uint2*>(&bt_lds[h0 + 1][d0]) = u;
                    u.x = pack_bf16(r0.z, r1.z); u.y = pack_bf16(r2.z, r3.z);
                    *reinterpret_cast<uint2*>(&bt_lds[h0 + 2][d0]) = u;
                    u.x = pack_bf16(r0.w, r1.w); u.y = pack_bf16(r2.w, r3.w);
                    *reinterpret_cast<uint2*>(&bt_lds[h0 + 3][d0]) = u;
                }
            }
            __syncthreads();
            // preload all fragments, then issue the 8-wmma burst back-to-back
            v16bf afrag[2], bfrag[4];
#pragma unroll
            for (int mt = 0; mt < 2; ++mt)
                afrag[mt] = load_frag_a(&a_lds[mt * 16 + lq][0], half);
#pragma unroll
            for (int nt = 0; nt < 4; ++nt)
                bfrag[nt] = load_frag_b(&bt_lds[wv * 64 + nt * 16 + lq][0], half);
#pragma unroll
            for (int nt = 0; nt < 4; ++nt) {
                acc1[0][nt] = wmma_bf16(afrag[0], bfrag[nt], acc1[0][nt]);
                acc1[1][nt] = wmma_bf16(afrag[1], bfrag[nt], acc1[1][nt]);
            }
            __syncthreads();
        }
    }

    // +b1, LayerNorm(512), ReLU -> h1bf
#pragma unroll
    for (int nt = 0; nt < 4; ++nt) {
        const float b1v = b1[t * 512 + wv * 64 + nt * 16 + lq];
#pragma unroll
        for (int mt = 0; mt < 2; ++mt)
#pragma unroll
            for (int j = 0; j < 8; ++j) acc1[mt][nt][j] += b1v;
    }
#pragma unroll
    for (int mt = 0; mt < 2; ++mt)
#pragma unroll
        for (int j = 0; j < 8; ++j) {
            float s = acc1[mt][0][j] + acc1[mt][1][j] + acc1[mt][2][j] + acc1[mt][3][j];
            float q = acc1[mt][0][j] * acc1[mt][0][j] + acc1[mt][1][j] * acc1[mt][1][j] +
                      acc1[mt][2][j] * acc1[mt][2][j] + acc1[mt][3][j] * acc1[mt][3][j];
#pragma unroll
            for (int m = 1; m <= 8; m <<= 1) { s += __shfl_xor(s, m, 32); q += __shfl_xor(q, m, 32); }
            if (lq == 0) { const int r = mt * 16 + half * 8 + j; psum[r][wv] = s; psq[r][wv] = q; }
        }
    __syncthreads();
    if (tid < 32) {
        float s = 0.f, q = 0.f;
#pragma unroll
        for (int w = 0; w < 8; ++w) { s += psum[tid][w]; q += psq[tid][w]; }
        const float mean = s * (1.f / 512.f);
        const float var  = q * (1.f / 512.f) - mean * mean;
        meanv[tid] = mean;
        rstdv[tid] = rsqrtf(var + 1e-5f);
    }
    __syncthreads();
#pragma unroll
    for (int nt = 0; nt < 4; ++nt) {
        const int col = wv * 64 + nt * 16 + lq;
        const float gv = g1[t * 512 + col];
        const float bv = be1[t * 512 + col];
#pragma unroll
        for (int mt = 0; mt < 2; ++mt)
#pragma unroll
            for (int j = 0; j < 8; ++j) {
                const int r = mt * 16 + half * 8 + j;
                float x = (acc1[mt][nt][j] - meanv[r]) * rstdv[r] * gv + bv;
                h1bf[r][col] = bf16_1(fmaxf(x, 0.f));
            }
    }
    __syncthreads();

    // ---------------- GEMM2: [32 x 512] x [512 x 256] ----------------
    v8f acc2[2][2];
#pragma unroll
    for (int mt = 0; mt < 2; ++mt) { acc2[mt][0] = zero8; acc2[mt][1] = zero8; }

#pragma unroll 1
    for (int kg = 0; kg < 512; kg += 32) {
        {
            const float* wbase = W2 + ((size_t)t * 512 + kg) * 256;
            if (kg + 32 < 512)    // prefetch next chunk (32KB)
                __builtin_prefetch(wbase + 32 * 256 + tid * 32, 0, 3);
#pragma unroll
            for (int it = 0; it < 2; ++it) {
                const int e  = it * 256 + tid;
                const int q  = e >> 6;          // 0..7 -> d0 = 4q
                const int h0 = (e & 63) * 4;
                const int d0 = 4 * q;
                const float* s0 = wbase + (size_t)d0 * 256 + h0;
                const float4 r0 = *reinterpret_cast<const float4*>(s0);
                const float4 r1 = *reinterpret_cast<const float4*>(s0 + 256);
                const float4 r2 = *reinterpret_cast<const float4*>(s0 + 512);
                const float4 r3 = *reinterpret_cast<const float4*>(s0 + 768);
                uint2 u;
                u.x = pack_bf16(r0.x, r1.x); u.y = pack_bf16(r2.x, r3.x);
                *reinterpret_cast<uint2*>(&bt_lds[h0 + 0][d0]) = u;
                u.x = pack_bf16(r0.y, r1.y); u.y = pack_bf16(r2.y, r3.y);
                *reinterpret_cast<uint2*>(&bt_lds[h0 + 1][d0]) = u;
                u.x = pack_bf16(r0.z, r1.z); u.y = pack_bf16(r2.z, r3.z);
                *reinterpret_cast<uint2*>(&bt_lds[h0 + 2][d0]) = u;
                u.x = pack_bf16(r0.w, r1.w); u.y = pack_bf16(r2.w, r3.w);
                *reinterpret_cast<uint2*>(&bt_lds[h0 + 3][d0]) = u;
            }
        }
        __syncthreads();
        v16bf afrag[2], bfrag[2];
#pragma unroll
        for (int mt = 0; mt < 2; ++mt)
            afrag[mt] = load_frag_a(&h1bf[mt * 16 + lq][kg], half);
#pragma unroll
        for (int nt = 0; nt < 2; ++nt)
            bfrag[nt] = load_frag_b(&bt_lds[wv * 32 + nt * 16 + lq][0], half);
#pragma unroll
        for (int nt = 0; nt < 2; ++nt) {
            acc2[0][nt] = wmma_bf16(afrag[0], bfrag[nt], acc2[0][nt]);
            acc2[1][nt] = wmma_bf16(afrag[1], bfrag[nt], acc2[1][nt]);
        }
        __syncthreads();
    }

    // +b2, LayerNorm(256), ReLU -> h2n (aliases bt_lds, now dead)
#pragma unroll
    for (int nt = 0; nt < 2; ++nt) {
        const float b2v = b2[t * 256 + wv * 32 + nt * 16 + lq];
#pragma unroll
        for (int mt = 0; mt < 2; ++mt)
#pragma unroll
            for (int j = 0; j < 8; ++j) acc2[mt][nt][j] += b2v;
    }
#pragma unroll
    for (int mt = 0; mt < 2; ++mt)
#pragma unroll
        for (int j = 0; j < 8; ++j) {
            float s = acc2[mt][0][j] + acc2[mt][1][j];
            float q = acc2[mt][0][j] * acc2[mt][0][j] + acc2[mt][1][j] * acc2[mt][1][j];
#pragma unroll
            for (int m = 1; m <= 8; m <<= 1) { s += __shfl_xor(s, m, 32); q += __shfl_xor(q, m, 32); }
            if (lq == 0) { const int r = mt * 16 + half * 8 + j; psum[r][wv] = s; psq[r][wv] = q; }
        }
    __syncthreads();
    if (tid < 32) {
        float s = 0.f, q = 0.f;
#pragma unroll
        for (int w = 0; w < 8; ++w) { s += psum[tid][w]; q += psq[tid][w]; }
        const float mean = s * (1.f / 256.f);
        const float var  = q * (1.f / 256.f) - mean * mean;
        meanv[tid] = mean;
        rstdv[tid] = rsqrtf(var + 1e-5f);
    }
    __syncthreads();
#pragma unroll
    for (int nt = 0; nt < 2; ++nt) {
        const int col = wv * 32 + nt * 16 + lq;
        const float gv = g2[t * 256 + col];
        const float bv = be2[t * 256 + col];
#pragma unroll
        for (int mt = 0; mt < 2; ++mt)
#pragma unroll
            for (int j = 0; j < 8; ++j) {
                const int r = mt * 16 + half * 8 + j;
                float x = (acc2[mt][nt][j] - meanv[r]) * rstdv[r] * gv + bv;
                h2n[r][col] = fmaxf(x, 0.f);
            }
    }
    // stage weight-net params (disjoint arrays; one barrier covers h2n too)
    {
        const float* wsrc = Ww + (size_t)t * 512 * 4;
        float* wwf = &wwl[0][0];
        for (int i = tid; i < 2048; i += 256) wwf[i] = wsrc[i];
        tel[tid] = temb[t * 256 + tid];
        if (tid < 4) bwl[tid] = bw[t * 4 + tid];
    }
    __syncthreads();

    // weight-net + softmax over M=4 within a lane quad
    if (tid < 128) {
        const int r = tid >> 2;
        const int m = tid & 3;
        float a = bwl[m];
#pragma unroll 8
        for (int h = 0; h < 256; ++h) a += h2n[r][h] * wwl[h][m];
#pragma unroll 8
        for (int h = 0; h < 256; ++h) a += tel[h] * wwl[256 + h][m];
        float mx = a;
        mx = fmaxf(mx, __shfl_xor(mx, 1, 32));
        mx = fmaxf(mx, __shfl_xor(mx, 2, 32));
        const float e = __expf(a - mx);
        float ssum = e;
        ssum += __shfl_xor(ssum, 1, 32);
        ssum += __shfl_xor(ssum, 2, 32);
        if (r < n_valid) out[(size_t)idx_sh[r] * 4 + m] = e / ssum;
    }
}

// ---------- launcher ----------
extern "C" void kernel_launch(void* const* d_in, const int* in_sizes, int n_in,
                              void* d_out, int out_size, void* d_ws, size_t ws_size,
                              hipStream_t stream) {
    (void)in_sizes; (void)n_in; (void)out_size; (void)ws_size;
    const float* f0   = (const float*)d_in[0];
    const float* f1   = (const float*)d_in[1];
    const float* f2   = (const float*)d_in[2];
    const float* f3   = (const float*)d_in[3];
    const float* W1   = (const float*)d_in[4];
    const float* b1   = (const float*)d_in[5];
    const float* g1   = (const float*)d_in[6];
    const float* be1  = (const float*)d_in[7];
    const float* W2   = (const float*)d_in[8];
    const float* b2   = (const float*)d_in[9];
    const float* g2   = (const float*)d_in[10];
    const float* be2  = (const float*)d_in[11];
    const float* temb = (const float*)d_in[12];
    const float* Ww   = (const float*)d_in[13];
    const float* bw   = (const float*)d_in[14];
    const int*   tids = (const int*)d_in[15];
    float* out = (float*)d_out;

    int* wsI   = (int*)d_ws;
    int* order = wsI;            // 4096
    int* count = wsI + 4096;     // 8
    int* off   = wsI + 4104;     // 8

    bucket_kernel<<<1, 256, 0, stream>>>(tids, order, count, off);
    fused_kernel<<<dim3(128, 8), 256, 0, stream>>>(f0, f1, f2, f3, W1, b1, g1, be1,
                                                   W2, b2, g2, be2, temb, Ww, bw,
                                                   order, count, off, out);
}
